// MetaLayer_618475290959
// MI455X (gfx1250) — matrix operand order repaired and checked by reading, not compiled, not run on hardware
//
#include <hip/hip_runtime.h>

typedef __attribute__((ext_vector_type(16))) __bf16 v16bf;
typedef __attribute__((ext_vector_type(8)))  float  v8f;

union FragB { uint4 q[2]; v16bf v; };

// ---------------------------------------------------------------------------
// Zero workspace (agg + deg) — harness poisons d_ws, so re-zero every launch.
// ---------------------------------------------------------------------------
__global__ void zero_ws_kernel(float* __restrict__ p, long n) {
  long i = (long)blockIdx.x * blockDim.x + threadIdx.x;
  long stride = (long)gridDim.x * blockDim.x;
  for (; i < n; i += stride) p[i] = 0.0f;
}

// ---------------------------------------------------------------------------
// Edge MLP: e = relu([f[r]|f[c]|ea] @ W1e + b1e) @ W2e + b2e
// plus scatter: agg[c] += e ; deg[c] += 1
// One wave handles a tile of 16 edges; WMMA bf16 16x16x32, f32 accumulate.
// ---------------------------------------------------------------------------
__global__ __launch_bounds__(256) void edge_mlp_kernel(
    const float* __restrict__ feats, const int* __restrict__ eidx,
    const float* __restrict__ eattr,
    const float* __restrict__ W1e, const float* __restrict__ b1e,
    const float* __restrict__ W2e, const float* __restrict__ b2e,
    float* __restrict__ e_out, float* __restrict__ agg, float* __restrict__ deg,
    int n_edges)
{
  __shared__ __bf16 w1[64 * 192];     // col-major: w1[n*192 + k], 24 KB
  __shared__ __bf16 w2[64 * 64];      // col-major: w2[n*64  + k],  8 KB
  __shared__ __bf16 hid[8][16 * 64];  // per-wave hidden tile,      16 KB

  // Stage weights as bf16, column-major (coalesced global reads).
  for (int i = threadIdx.x; i < 192 * 64; i += blockDim.x) {
    int k = i >> 6, n = i & 63;
    w1[n * 192 + k] = (__bf16)W1e[i];
  }
  for (int i = threadIdx.x; i < 64 * 64; i += blockDim.x) {
    int k = i >> 6, n = i & 63;
    w2[n * 64 + k] = (__bf16)W2e[i];
  }
  __syncthreads();

  const int lane = threadIdx.x & 31;
  const int wid  = threadIdx.x >> 5;
  const int row  = lane & 15;   // M row owned by this lane
  const int hb   = lane >> 4;   // lane half (A-layout K interleave)
  const int nwaves = gridDim.x * (blockDim.x >> 5);
  int wave = blockIdx.x * (blockDim.x >> 5) + wid;
  const int ntiles = n_edges >> 4;   // 800000/16 = 50000, exact
  __bf16* hbuf = hid[wid];

  for (int tile = wave; tile < ntiles; tile += nwaves) {
    const int e0  = tile << 4;
    const int myE = e0 + row;
    const int rI  = eidx[2 * myE + 0];
    const int cI  = eidx[2 * myE + 1];
    if (hb == 0) atomicAdd(&deg[cI], 1.0f);   // one count per edge

    const float* fr = feats + (long)rI * 64;
    const float* fc = feats + (long)cI * 64;
    const float* ea = eattr + (long)myE * 64;
    // x row = [ f[r] (0..63) | f[c] (64..127) | ea (128..191) ]
    auto srcp = [&](int k) -> const float* {
      return k < 64 ? (fr + k) : (k < 128 ? (fc + (k - 64)) : (ea + (k - 128)));
    };

    // -------- layer 1: K = 192 (6 k-steps), N = 64 (4 n-tiles) --------
    v8f acc[4] = {};
    for (int s = 0; s < 6; ++s) {
      const int k0 = s * 32 + hb * 8;
      const float* p0 = srcp(k0);
      const float* p1 = srcp(k0 + 16);
      float4 q0 = *(const float4*)(p0);
      float4 q1 = *(const float4*)(p0 + 4);
      float4 q2 = *(const float4*)(p1);
      float4 q3 = *(const float4*)(p1 + 4);
      v16bf a;
      a[0]=(__bf16)q0.x; a[1]=(__bf16)q0.y; a[2]=(__bf16)q0.z; a[3]=(__bf16)q0.w;
      a[4]=(__bf16)q1.x; a[5]=(__bf16)q1.y; a[6]=(__bf16)q1.z; a[7]=(__bf16)q1.w;
      a[8]=(__bf16)q2.x; a[9]=(__bf16)q2.y; a[10]=(__bf16)q2.z; a[11]=(__bf16)q2.w;
      a[12]=(__bf16)q3.x; a[13]=(__bf16)q3.y; a[14]=(__bf16)q3.z; a[15]=(__bf16)q3.w;

      const int kb = s * 32 + hb * 8;    // B layout: K interleaved like A (8 at kb, 8 at kb+16)
      for (int t = 0; t < 4; ++t) {
        FragB B;
        const __bf16* wp = &w1[(t * 16 + row) * 192 + kb];
        B.q[0] = *(const uint4*)(wp);
        B.q[1] = *(const uint4*)(wp + 16);
        acc[t] = __builtin_amdgcn_wmma_f32_16x16x32_bf16(
            false, a, false, B.v, (short)0, acc[t], false, false);
      }
    }

    // bias + ReLU -> per-wave LDS hidden tile (bf16, row-major 16x64)
    for (int t = 0; t < 4; ++t) {
      const int n = t * 16 + row;
      const float bias = b1e[n];
      for (int v = 0; v < 8; ++v) {
        const int m = v + hb * 8;                 // C layout: VGPR v -> M
        float x = acc[t][v] + bias;
        hbuf[m * 64 + n] = (__bf16)(x > 0.0f ? x : 0.0f);
      }
    }
    __builtin_amdgcn_wave_barrier();  // per-wave LDS ops are in-order (DScnt)

    // -------- layer 2: K = 64 (2 k-steps), N = 64 --------
    v8f acc2[4] = {};
    for (int s = 0; s < 2; ++s) {
      FragB A;
      const __bf16* hp = &hbuf[row * 64 + s * 32 + hb * 8];
      A.q[0] = *(const uint4*)(hp);
      A.q[1] = *(const uint4*)(hp + 16);
      const int kb = s * 32 + hb * 8;
      for (int t = 0; t < 4; ++t) {
        FragB B;
        const __bf16* wp = &w2[(t * 16 + row) * 64 + kb];
        B.q[0] = *(const uint4*)(wp);
        B.q[1] = *(const uint4*)(wp + 16);
        acc2[t] = __builtin_amdgcn_wmma_f32_16x16x32_bf16(
            false, A.v, false, B.v, (short)0, acc2[t], false, false);
      }
    }

    // column index of each output row m (broadcast from the lane owning it)
    int cM[8];
    for (int v = 0; v < 8; ++v) cM[v] = __shfl(cI, v + hb * 8, 32);

    float* ebase = e_out + (long)e0 * 64;
    for (int t = 0; t < 4; ++t) {
      const int n = t * 16 + row;
      const float bias = b2e[n];
      for (int v = 0; v < 8; ++v) {
        const int m = v + hb * 8;
        const float val = acc2[t][v] + bias;
        ebase[m * 64 + n] = val;
        atomicAdd(&agg[(long)cM[v] * 64 + n], val);
      }
    }
  }
}

// ---------------------------------------------------------------------------
// Node MLP: out = relu([feats | agg/max(deg,1)] @ W1n + b1n) @ W2n + b2n
// ---------------------------------------------------------------------------
__global__ __launch_bounds__(256) void node_mlp_kernel(
    const float* __restrict__ feats, const float* __restrict__ agg,
    const float* __restrict__ deg,
    const float* __restrict__ W1n, const float* __restrict__ b1n,
    const float* __restrict__ W2n, const float* __restrict__ b2n,
    float* __restrict__ out, int n_nodes)
{
  __shared__ __bf16 w1[64 * 128];     // col-major, 16 KB
  __shared__ __bf16 w2[64 * 64];      // col-major,  8 KB
  __shared__ __bf16 hid[8][16 * 64];  // 16 KB

  for (int i = threadIdx.x; i < 128 * 64; i += blockDim.x) {
    int k = i >> 6, n = i & 63;
    w1[n * 128 + k] = (__bf16)W1n[i];
  }
  for (int i = threadIdx.x; i < 64 * 64; i += blockDim.x) {
    int k = i >> 6, n = i & 63;
    w2[n * 64 + k] = (__bf16)W2n[i];
  }
  __syncthreads();

  const int lane = threadIdx.x & 31;
  const int wid  = threadIdx.x >> 5;
  const int row  = lane & 15;
  const int hb   = lane >> 4;
  const int nwaves = gridDim.x * (blockDim.x >> 5);
  int wave = blockIdx.x * (blockDim.x >> 5) + wid;
  const int ntiles = n_nodes >> 4;   // 50000/16 = 3125, exact
  __bf16* hbuf = hid[wid];

  for (int tile = wave; tile < ntiles; tile += nwaves) {
    const int node = (tile << 4) + row;
    const float d  = deg[node];
    const float inv = 1.0f / fmaxf(d, 1.0f);
    const float* fsrc = feats + (long)node * 64;
    const float* gsrc = agg   + (long)node * 64;

    // h row = [ feats (0..63) | agg*inv (64..127) ]
    auto lda8 = [&](int k, int off, v16bf& a) {
      const float* p; float sc;
      if (k < 64) { p = fsrc + k;        sc = 1.0f; }
      else        { p = gsrc + (k - 64); sc = inv;  }
      float4 q0 = *(const float4*)(p);
      float4 q1 = *(const float4*)(p + 4);
      a[off+0]=(__bf16)(q0.x*sc); a[off+1]=(__bf16)(q0.y*sc);
      a[off+2]=(__bf16)(q0.z*sc); a[off+3]=(__bf16)(q0.w*sc);
      a[off+4]=(__bf16)(q1.x*sc); a[off+5]=(__bf16)(q1.y*sc);
      a[off+6]=(__bf16)(q1.z*sc); a[off+7]=(__bf16)(q1.w*sc);
    };

    // -------- layer 1: K = 128 (4 k-steps) --------
    v8f acc[4] = {};
    for (int s = 0; s < 4; ++s) {
      const int k0 = s * 32 + hb * 8;
      v16bf a;
      lda8(k0,      0, a);
      lda8(k0 + 16, 8, a);
      const int kb = s * 32 + hb * 8;
      for (int t = 0; t < 4; ++t) {
        FragB B;
        const __bf16* wp = &w1[(t * 16 + row) * 128 + kb];
        B.q[0] = *(const uint4*)(wp);
        B.q[1] = *(const uint4*)(wp + 16);
        acc[t] = __builtin_amdgcn_wmma_f32_16x16x32_bf16(
            false, a, false, B.v, (short)0, acc[t], false, false);
      }
    }

    for (int t = 0; t < 4; ++t) {
      const int n = t * 16 + row;
      const float bias = b1n[n];
      for (int v = 0; v < 8; ++v) {
        const int m = v + hb * 8;
        float x = acc[t][v] + bias;
        hbuf[m * 64 + n] = (__bf16)(x > 0.0f ? x : 0.0f);
      }
    }
    __builtin_amdgcn_wave_barrier();

    // -------- layer 2: K = 64 (2 k-steps) --------
    v8f acc2[4] = {};
    for (int s = 0; s < 2; ++s) {
      FragB A;
      const __bf16* hp = &hbuf[row * 64 + s * 32 + hb * 8];
      A.q[0] = *(const uint4*)(hp);
      A.q[1] = *(const uint4*)(hp + 16);
      const int kb = s * 32 + hb * 8;
      for (int t = 0; t < 4; ++t) {
        FragB B;
        const __bf16* wp = &w2[(t * 16 + row) * 64 + kb];
        B.q[0] = *(const uint4*)(wp);
        B.q[1] = *(const uint4*)(wp + 16);
        acc2[t] = __builtin_amdgcn_wmma_f32_16x16x32_bf16(
            false, A.v, false, B.v, (short)0, acc2[t], false, false);
      }
    }

    float* obase = out + (long)(tile << 4) * 64;
    for (int t = 0; t < 4; ++t) {
      const int n = t * 16 + row;
      const float bias = b2n[n];
      for (int v = 0; v < 8; ++v) {
        const int m = v + hb * 8;
        obase[m * 64 + n] = acc2[t][v] + bias;
      }
    }
  }
}

// ---------------------------------------------------------------------------
extern "C" void kernel_launch(void* const* d_in, const int* in_sizes, int n_in,
                              void* d_out, int out_size, void* d_ws, size_t ws_size,
                              hipStream_t stream) {
  const float* feats = (const float*)d_in[0];
  const int*   eidx  = (const int*)d_in[1];
  const float* eattr = (const float*)d_in[2];
  const float* W1e   = (const float*)d_in[3];
  const float* b1e   = (const float*)d_in[4];
  const float* W2e   = (const float*)d_in[5];
  const float* b2e   = (const float*)d_in[6];
  const float* W1n   = (const float*)d_in[7];
  const float* b1n   = (const float*)d_in[8];
  const float* W2n   = (const float*)d_in[9];
  const float* b2n   = (const float*)d_in[10];

  const int n_nodes = in_sizes[0] / 64;
  const int n_edges = in_sizes[2] / 64;

  float* feats_out = (float*)d_out;
  float* e_out     = feats_out + (size_t)n_nodes * 64;

  float* agg = (float*)d_ws;                       // [n_nodes * 64]
  float* deg = agg + (size_t)n_nodes * 64;         // [n_nodes]

  const long zn = (long)n_nodes * 64 + n_nodes;
  zero_ws_kernel<<<512, 256, 0, stream>>>(agg, zn);

  edge_mlp_kernel<<<1024, 256, 0, stream>>>(feats, eidx, eattr,
                                            W1e, b1e, W2e, b2e,
                                            e_out, agg, deg, n_edges);

  node_mlp_kernel<<<256, 256, 0, stream>>>(feats, agg, deg,
                                           W1n, b1n, W2n, b2n,
                                           feats_out, n_nodes);
}
